// SpatialSTEncoder_8100308320612
// MI455X (gfx1250) — matrix operand order repaired and checked by reading, not compiled, not run on hardware
//
#include <hip/hip_runtime.h>
#include <hip/hip_bf16.h>

// ---------------------------------------------------------------------------
// SpatialSTEncoder forward for gfx1250 (MI455X), wave32 + WMMA f16->f32.
// Assumes d_in[] is the recursive insertion-order flatten of setup_inputs():
//  0 expr, 1 coords,
//  2 gene_emb, 3 gene_pos, 4 val_w, 5 val_b, 6 val_g, 7 val_beta,
//  8 sp_w, 9 sp_b, 10 sp_g, 11 sp_beta,
//  12.. per layer (x2): ln1_g,ln1_b,wq,bq,wk,bk,wv,bv,wo,bo,ln2_g,ln2_b,w1,b1,w2,b2
//  44 q_w,45 q_b,46 k_w,47 k_b,48 v_w,49 v_b,50 o_w,51 o_b, 52 proj
//
// All GEMM shapes are padded so that every tile is full (R%64, K%32, N%64 == 0):
// no bounds checks anywhere in the GEMM -> pure b128 staging + WMMA inner loop.
// A-tile staging optionally uses the gfx1250 async global->LDS copy path.
// ---------------------------------------------------------------------------

typedef _Float16 f16;
typedef _Float16 v16h __attribute__((ext_vector_type(16)));
typedef float    v8f  __attribute__((ext_vector_type(8)));

#ifdef __has_builtin
#  if __has_builtin(__builtin_amdgcn_global_load_async_to_lds_b128)
#    define HAVE_ASYNC_LDS 1
#  else
#    define HAVE_ASYNC_LDS 0
#  endif
#else
#  define HAVE_ASYNC_LDS 0
#endif

#if HAVE_ASYNC_LDS
typedef int vi4 __attribute__((vector_size(16)));
typedef __attribute__((address_space(1))) vi4* gvi4p;   // global int4*
typedef __attribute__((address_space(3))) vi4* lvi4p;   // LDS int4*
#endif

#define Nn     256
#define Gg     2000
#define TOPKk  512
#define Dd     256
#define Hh     4
#define DHd    64
#define DEPTHn 2
#define FFf    1024
#define Mm     266
#define Mp     320             // padded random-feature dim (5*64)
#define SVALID 513
#define SPAD   576             // padded sequence length (9*64)
#define NCC    16              // cells per chunk
#define RCC    (NCC*SPAD)      // 9216 rows per chunk

enum { OM_F16 = 0, OM_F32 = 1, OM_ACC = 2 };
enum { FG_GELU = 1, FG_ZPAD = 2 };

__device__ __forceinline__ float gelu_f(float v) {
  return 0.5f * v * (1.f + erff(v * 0.70710678118f));
}

// ---------------------------------------------------------------------------
// Tiled WMMA GEMM: C[b] = A[b] (R x K) * B[b] (K x Ncols)  (+bias, epilogues)
// Block tile 64x64, TK=32, 256 threads = 8 waves (2 row-groups x 4 col-groups).
// Each wave computes a 32x16 slab = 2 WMMAs per K-step sharing one B fragment.
// Shapes are full by construction: unguarded contiguous 16B loads (b128).
// ATRANS: A stored [K][lda] (kp^T * v), strided scalar gather.
// Batch: blockIdx.z -> (bn = z/Hdiv, bh = z%Hdiv) with per-dim strides.
// ---------------------------------------------------------------------------
template <bool ATRANS>
__global__ __launch_bounds__(256) void k_gemm(
    const f16* __restrict__ A, int lda, long sAn, long sAh,
    const f16* __restrict__ B, int ldb, long sBn, long sBh,
    const float* __restrict__ bias,
    void* __restrict__ Cout, int outmode, int ldc, long sCn, long sCh,
    const float* __restrict__ rs, long sRn, long sRh,
    int Kk, int Hdiv, int flags)
{
  const int z = blockIdx.z, bn = z / Hdiv, bh = z % Hdiv;
  const f16* Ab = A + (long)bn * sAn + (long)bh * sAh;
  const f16* Bb = B + (long)bn * sBn + (long)bh * sBh;
  const long coff = (long)bn * sCn + (long)bh * sCh;
  const int row0 = blockIdx.y * 64;
  const int col0 = blockIdx.x * 64;
  __shared__ __attribute__((aligned(16))) f16 As[64][40];
  __shared__ __attribute__((aligned(16))) f16 Bs[64][40];   // transposed: Bs[n][k]
  const int tid = threadIdx.x;
  const int w = tid >> 5, lane = tid & 31;
  const int wr = w >> 2, wc = w & 3;
  const int fm = lane & 15, hk = lane >> 4;
  // staging thread mapping
  const int ar = tid >> 2, ak = (tid & 3) << 3;    // A: 64 rows x 32k, 8 f16/thread
  const int bk = tid >> 3, bn0 = (tid & 7) << 3;   // B: 32k x 64n,   8 f16/thread
  v8f acc0 = {}, acc1 = {};
  for (int k0 = 0; k0 < Kk; k0 += 32) {
    __syncthreads();
    if (!ATRANS) {
      const f16* ap = Ab + (long)(row0 + ar) * lda + (k0 + ak);
      if (k0 + 32 < Kk) __builtin_prefetch(ap + 32, 0, 3);
#if HAVE_ASYNC_LDS
      // gfx1250 async copy: global -> LDS, 16B per lane, tracked by ASYNCcnt
      __builtin_amdgcn_global_load_async_to_lds_b128(
          (gvi4p)(vi4*)ap, (lvi4p)(vi4*)&As[ar][ak], 0, 0);
#else
#pragma unroll
      for (int i = 0; i < 8; ++i) As[ar][ak + i] = ap[i];            // b128 ld/st
#endif
    } else {
      const f16* ap = Ab + (long)(k0 + ak) * lda + (row0 + ar);
#pragma unroll
      for (int i = 0; i < 8; ++i) As[ar][ak + i] = ap[(long)i * lda];
    }
    {
      const f16* bp = Bb + (long)(k0 + bk) * ldb + (col0 + bn0);
      if (k0 + 32 < Kk) __builtin_prefetch(bp + (long)32 * ldb, 0, 3);
#pragma unroll
      for (int i = 0; i < 8; ++i) Bs[bn0 + i][bk] = bp[i];           // b128 ld
    }
#if HAVE_ASYNC_LDS
    if (!ATRANS) {
#  if __has_builtin(__builtin_amdgcn_s_wait_asynccnt)
      __builtin_amdgcn_s_wait_asynccnt(0);
#  else
      asm volatile("s_wait_asynccnt 0x0" ::: "memory");
#  endif
    }
#endif
    __syncthreads();
    // Fragments per ISA 7.12.2 (wave32):
    // A: lane fm = row M, elems 0..7 -> K=8*hk+j, 8..15 -> K=16+8*hk+j
    // B: lane fm = col N, elems 0..7 -> K=8*hk+j, 8..15 -> K=16+8*hk+j
    v16h a0, a1, b;
    const f16* Br = &Bs[wc * 16 + fm][0];
#pragma unroll
    for (int j = 0; j < 8; ++j) { b[j] = Br[hk * 8 + j]; b[8 + j] = Br[16 + hk * 8 + j]; }
    const f16* Ar0 = &As[wr * 32 + fm][0];
    const f16* Ar1 = &As[wr * 32 + 16 + fm][0];
#pragma unroll
    for (int j = 0; j < 8; ++j) { a0[j] = Ar0[hk * 8 + j]; a0[8 + j] = Ar0[16 + hk * 8 + j]; }
#pragma unroll
    for (int j = 0; j < 8; ++j) { a1[j] = Ar1[hk * 8 + j]; a1[8 + j] = Ar1[16 + hk * 8 + j]; }
    acc0 = __builtin_amdgcn_wmma_f32_16x16x32_f16(false, a0, false, b, (short)0, acc0,
                                                  false, false);
    acc1 = __builtin_amdgcn_wmma_f32_16x16x32_f16(false, a1, false, b, (short)0, acc1,
                                                  false, false);
  }
  const int col = col0 + wc * 16 + fm;
  const float bv = bias ? bias[col] : 0.f;
#pragma unroll
  for (int sub = 0; sub < 2; ++sub) {
    const v8f acc = sub ? acc1 : acc0;
#pragma unroll
    for (int r = 0; r < 8; ++r) {
      const int row = row0 + wr * 32 + sub * 16 + hk * 8 + r;  // C: VGPR r -> M=r+8*hk
      float v;
      if ((flags & FG_ZPAD) && (row % SPAD) >= SVALID) {
        v = 0.f;
      } else {
        v = acc[r] + bv;
        if (flags & FG_GELU) v = gelu_f(v);
        if (rs) v *= rs[(long)bn * sRn + (long)bh * sRh + row];
      }
      const long ci = coff + (long)row * ldc + col;
      if (outmode == OM_F16)      ((f16*)Cout)[ci] = (f16)v;
      else if (outmode == OM_F32) ((float*)Cout)[ci] = v;
      else                        ((float*)Cout)[ci] += v;   // residual accumulate
    }
  }
}

// ---------------------------------------------------------------------------
// Deterministic top-512-of-2000 per cell (order-free set selection via float
// threshold bisection + block prefix scan; no cross-block atomics).
// ---------------------------------------------------------------------------
__global__ __launch_bounds__(256) void k_topk(const float* __restrict__ expr,
                                              float* __restrict__ vals,
                                              int* __restrict__ idxb, int cellBase)
{
  const int n = cellBase + blockIdx.x;
  const int t = threadIdx.x;
  __shared__ float sv[Gg];
  __shared__ float red[256];
  __shared__ int   ip[256];
  __shared__ float slo, shi;
  __shared__ int   scnt;
  for (int i = t; i < Gg; i += 256) sv[i] = expr[(long)n * Gg + i];
  __syncthreads();
  float mn = 1e30f, mx = -1e30f;
  for (int i = t; i < Gg; i += 256) { float v = sv[i]; mn = fminf(mn, v); mx = fmaxf(mx, v); }
  red[t] = mx; __syncthreads();
  for (int o = 128; o > 0; o >>= 1) { if (t < o) red[t] = fmaxf(red[t], red[t + o]); __syncthreads(); }
  if (t == 0) shi = red[0] + 1.f;
  __syncthreads();
  red[t] = mn; __syncthreads();
  for (int o = 128; o > 0; o >>= 1) { if (t < o) red[t] = fminf(red[t], red[t + o]); __syncthreads(); }
  if (t == 0) { slo = red[0] - 1.f; scnt = 0; }
  __syncthreads();
  for (int it = 0; it < 64; ++it) {
    const float mid = 0.5f * (slo + shi);
    int c = 0;
    for (int i = t; i < Gg; i += 256) c += (sv[i] > mid) ? 1 : 0;
    atomicAdd(&scnt, c);
    __syncthreads();
    if (t == 0) { if (scnt > TOPKk) slo = mid; else shi = mid; scnt = 0; }
    __syncthreads();
  }
  const float thr = shi, lo = slo;
  // pass 1: strictly above threshold (deterministic positions via scan)
  int c1 = 0;
  for (int i = t; i < Gg; i += 256) c1 += (sv[i] > thr) ? 1 : 0;
  ip[t] = c1; __syncthreads();
  for (int o = 1; o < 256; o <<= 1) {
    int v = (t >= o) ? ip[t - o] : 0; __syncthreads();
    ip[t] += v; __syncthreads();
  }
  const int total1 = ip[255];
  int p = ip[t] - c1;
  for (int i = t; i < Gg; i += 256)
    if (sv[i] > thr) { vals[(long)n * TOPKk + p] = sv[i]; idxb[(long)n * TOPKk + p] = i; ++p; }
  __syncthreads();
  // pass 2: boundary band (lo, thr], fill up to 512
  int c2 = 0;
  for (int i = t; i < Gg; i += 256) c2 += (sv[i] > lo && sv[i] <= thr) ? 1 : 0;
  ip[t] = c2; __syncthreads();
  for (int o = 1; o < 256; o <<= 1) {
    int v = (t >= o) ? ip[t - o] : 0; __syncthreads();
    ip[t] += v; __syncthreads();
  }
  p = total1 + ip[t] - c2;
  for (int i = t; i < Gg; i += 256)
    if (sv[i] > lo && sv[i] <= thr) {
      if (p < TOPKk) { vals[(long)n * TOPKk + p] = sv[i]; idxb[(long)n * TOPKk + p] = i; }
      ++p;
    }
}

// ---------------------------------------------------------------------------
// Token embedding: x[nl, srow, :] (chunk-local fp32), zero padded rows.
// ---------------------------------------------------------------------------
__global__ __launch_bounds__(256) void k_embed(
    const float* __restrict__ coords,
    const float* sp_w, const float* sp_b, const float* sp_g, const float* sp_be,
    const float* val_w, const float* val_b, const float* val_g, const float* val_be,
    const float* __restrict__ gene_emb, const float* __restrict__ gene_pos,
    const float* __restrict__ vals, const int* __restrict__ idxb,
    float* __restrict__ x, int cellBase)
{
  const int srow = blockIdx.x, nl = blockIdx.y, n = cellBase + nl, t = threadIdx.x;
  const long orow = ((long)nl * SPAD + srow) * Dd;
  if (srow >= SVALID) { x[orow + t] = 0.f; return; }
  __shared__ float red[256];
  float pre, gg, bb, emb = 0.f;
  if (srow == 0) {
    const float c0 = coords[n * 2], c1 = coords[n * 2 + 1];
    pre = c0 * sp_w[t] + c1 * sp_w[Dd + t] + sp_b[t];
    gg = sp_g[t]; bb = sp_be[t];
  } else {
    const int j = srow - 1;
    const float v = vals[(long)n * TOPKk + j];
    const int gi = idxb[(long)n * TOPKk + j];
    pre = v * val_w[t] + val_b[t];
    gg = val_g[t]; bb = val_be[t];
    emb = gene_emb[(long)gi * Dd + t] + gene_pos[(long)gi * Dd + t];
  }
  red[t] = pre; __syncthreads();
  for (int o = 128; o > 0; o >>= 1) { if (t < o) red[t] += red[t + o]; __syncthreads(); }
  const float mu = red[0] * (1.f / Dd); __syncthreads();
  const float c = pre - mu;
  red[t] = c * c; __syncthreads();
  for (int o = 128; o > 0; o >>= 1) { if (t < o) red[t] += red[t + o]; __syncthreads(); }
  const float var = red[0] * (1.f / Dd);
  x[orow + t] = emb + gelu_f(c * rsqrtf(var + 1e-5f) * gg + bb);
}

// LayerNorm row -> f16 (zero padded rows)
__global__ __launch_bounds__(256) void k_ln(const float* __restrict__ x,
                                            const float* __restrict__ g,
                                            const float* __restrict__ b,
                                            f16* __restrict__ out)
{
  const int row = blockIdx.x, t = threadIdx.x;
  if ((row % SPAD) >= SVALID) { out[(long)row * Dd + t] = (f16)0.f; return; }
  __shared__ float red[256];
  const float v = x[(long)row * Dd + t];
  red[t] = v; __syncthreads();
  for (int o = 128; o > 0; o >>= 1) { if (t < o) red[t] += red[t + o]; __syncthreads(); }
  const float mu = red[0] * (1.f / Dd); __syncthreads();
  const float c = v - mu;
  red[t] = c * c; __syncthreads();
  for (int o = 128; o > 0; o >>= 1) { if (t < o) red[t] += red[t + o]; __syncthreads(); }
  const float var = red[0] * (1.f / Dd);
  out[(long)row * Dd + t] = (f16)(c * rsqrtf(var + 1e-5f) * g[t] + b[t]);
}

// diag = 0.5 * sum_d x^2 * dn^2 per (row, head), dn^2 = 1/8, for q and k
__global__ __launch_bounds__(256) void k_diag(const f16* __restrict__ q16,
                                              const f16* __restrict__ k16,
                                              float* __restrict__ diagq,
                                              float* __restrict__ diagk)
{
  const int row = blockIdx.x, t = threadIdx.x, h = t >> 6, d = t & 63;
  __shared__ float red[256];
  float qv = (float)q16[(long)row * Dd + t];
  red[t] = qv * qv; __syncthreads();
  for (int o = 32; o > 0; o >>= 1) { if (d < o) red[t] += red[t + o]; __syncthreads(); }
  if (d == 0) diagq[(long)h * RCC + row] = red[t] * 0.0625f;
  __syncthreads();
  float kv = (float)k16[(long)row * Dd + t];
  red[t] = kv * kv; __syncthreads();
  for (int o = 32; o > 0; o >>= 1) { if (d < o) red[t] += red[t + o]; __syncthreads(); }
  if (d == 0) diagk[(long)h * RCC + row] = red[t] * 0.0625f;
}

// per-(row,h) max over m<266 of xd (query stabilizer)
__global__ __launch_bounds__(256) void k_rowmax(const float* __restrict__ xdf,
                                                float* __restrict__ stabq)
{
  const int row = blockIdx.x, h = blockIdx.y, t = threadIdx.x;
  const float* src = xdf + ((long)h * RCC + row) * Mp;
  float mx = -1e30f;
  for (int i = t; i < Mm; i += 256) mx = fmaxf(mx, src[i]);
  __shared__ float red[256];
  red[t] = mx; __syncthreads();
  for (int o = 128; o > 0; o >>= 1) { if (t < o) red[t] = fmaxf(red[t], red[t + o]); __syncthreads(); }
  if (t == 0) stabq[(long)h * RCC + row] = red[0];
}

// per-(cell,h) max over valid s<513, m<266 (key stabilizer)
__global__ __launch_bounds__(256) void k_headmax(const float* __restrict__ xdf,
                                                 float* __restrict__ stabk)
{
  const int nl = blockIdx.x, h = blockIdx.y, t = threadIdx.x;
  float mx = -1e30f;
  for (int s = 0; s < SVALID; ++s) {
    const float* src = xdf + ((long)h * RCC + (long)nl * SPAD + s) * Mp;
    for (int i = t; i < Mm; i += 256) mx = fmaxf(mx, src[i]);
  }
  __shared__ float red[256];
  red[t] = mx; __syncthreads();
  for (int o = 128; o > 0; o >>= 1) { if (t < o) red[t] = fmaxf(red[t], red[t + o]); __syncthreads(); }
  if (t == 0) stabk[nl * Hh + h] = red[0];
}

// qp/kp = ratio * (exp(xd - diag - stab) + eps), masked pads
__global__ __launch_bounds__(256) void k_exp(const float* __restrict__ xdf,
                                             const float* __restrict__ diag,
                                             const float* __restrict__ stab,
                                             f16* __restrict__ dst, int isq)
{
  const int row = blockIdx.x, h = blockIdx.y, t = threadIdx.x;
  const long base = ((long)h * RCC + row) * Mp;
  const float dg = diag[(long)h * RCC + row];
  const float st = isq ? stab[(long)h * RCC + row] : stab[(row / SPAD) * Hh + h];
  const int srow = row % SPAD;
  for (int m = t; m < Mp; m += 256) {
    float o;
    if (m >= Mm || (!isq && srow >= SVALID)) o = 0.f;
    else o = 0.06131393f * (expf(xdf[base + m] - dg - st) + 1e-4f);   // ratio = 266^-0.5
    dst[base + m] = (f16)o;
  }
}

// kpsum[m] = sum_s kp[s,m] per (cell, head)
__global__ __launch_bounds__(256) void k_kpsum(const f16* __restrict__ kp,
                                               float* __restrict__ kpsum)
{
  const int nl = blockIdx.x, h = blockIdx.y, t = threadIdx.x;
  for (int m = t; m < Mp; m += 256) {
    float a = 0.f;
    const f16* base = kp + ((long)h * RCC + (long)nl * SPAD) * Mp + m;
    for (int s = 0; s < SPAD; ++s) a += (float)base[(long)s * Mp];
    kpsum[((long)nl * Hh + h) * Mp + m] = a;
  }
}

// d_inv[row,h] = 1 / (qp[row,:] . kpsum[:])
__global__ __launch_bounds__(256) void k_dinv(const f16* __restrict__ qp,
                                              const float* __restrict__ kpsum,
                                              float* __restrict__ dinv)
{
  const int row = blockIdx.x, t = threadIdx.x, h = t >> 6, d = t & 63;
  const int nl = row / SPAD;
  const f16* qrow = qp + ((long)h * RCC + row) * Mp;
  const float* ks = kpsum + ((long)nl * Hh + h) * Mp;
  float a = 0.f;
  for (int m = d; m < Mp; m += 64) a += (float)qrow[m] * ks[m];
  __shared__ float red[256];
  red[t] = a; __syncthreads();
  for (int o = 32; o > 0; o >>= 1) { if (d < o) red[t] += red[t + o]; __syncthreads(); }
  if (d == 0) dinv[(long)h * RCC + row] = 1.f / red[t];
}

__global__ __launch_bounds__(256) void k_tof16(const float* __restrict__ x,
                                               f16* __restrict__ o, long nelem)
{
  const long i = (long)blockIdx.x * 256 + threadIdx.x;
  if (i < nelem) o[i] = (f16)x[i];
}

__global__ __launch_bounds__(256) void k_cvt(const float* __restrict__ s,
                                             f16* __restrict__ d, long n)
{
  const long i = (long)blockIdx.x * 256 + threadIdx.x;
  if (i < n) d[i] = (f16)s[i];
}

// pmT[d][m] = dn * pm[m][d] (dn = 64^-0.25 folded in), zero-padded m >= 266
__global__ __launch_bounds__(256) void k_pmT(const float* __restrict__ pm,
                                             f16* __restrict__ dst)
{
  const int id = blockIdx.x * 256 + threadIdx.x;
  if (id >= DHd * Mp) return;
  const int dd = id / Mp, mm = id % Mp;
  const float v = (mm < Mm) ? pm[mm * DHd + dd] * 0.35355339059f : 0.f;
  dst[id] = (f16)v;
}

// q = x[n,0,:] @ q_w + q_b (fp32, spatial token only)
__global__ __launch_bounds__(256) void k_qpool(const float* __restrict__ x,
                                               const float* __restrict__ qw,
                                               const float* __restrict__ qb,
                                               float* __restrict__ qpool, int cellBase)
{
  const int nl = blockIdx.x, n = cellBase + nl, t = threadIdx.x;
  __shared__ float xr[256];
  xr[t] = x[(long)nl * SPAD * Dd + t];
  __syncthreads();
  float acc = qb[t];
  for (int kk = 0; kk < Dd; ++kk) acc += xr[kk] * qw[kk * Dd + t];
  qpool[(long)n * Dd + t] = acc;
}

// softmax cross-attention pooling per cell
__global__ __launch_bounds__(256) void k_pool(const f16* __restrict__ kk,
                                              const f16* __restrict__ vv,
                                              const float* __restrict__ qpool,
                                              f16* __restrict__ pooled, int cellBase)
{
  const int nl = blockIdx.x, n = cellBase + nl, t = threadIdx.x;
  __shared__ float qv[256];
  __shared__ float lg[512];
  __shared__ float red[256];
  qv[t] = qpool[(long)n * Dd + t];
  __syncthreads();
  for (int j = t; j < 512; j += 256) {
    const f16* kr = kk + ((long)nl * SPAD + 1 + j) * Dd;
    float a = 0.f;
    for (int d = 0; d < Dd; ++d) a += qv[d] * (float)kr[d];
    lg[j] = a * 0.0625f;   // 1/sqrt(D)
  }
  __syncthreads();
  red[t] = fmaxf(lg[t], lg[t + 256]); __syncthreads();
  for (int o = 128; o > 0; o >>= 1) { if (t < o) red[t] = fmaxf(red[t], red[t + o]); __syncthreads(); }
  const float mx = red[0]; __syncthreads();
  const float e1 = expf(lg[t] - mx), e2 = expf(lg[t + 256] - mx);
  lg[t] = e1; lg[t + 256] = e2;
  red[t] = e1 + e2; __syncthreads();
  for (int o = 128; o > 0; o >>= 1) { if (t < o) red[t] += red[t + o]; __syncthreads(); }
  const float inv = 1.f / red[0];
  __syncthreads();
  float acc = 0.f;
  for (int j = 0; j < 512; ++j)
    acc += lg[j] * (float)vv[((long)nl * SPAD + 1 + j) * Dd + t];
  pooled[(long)n * Dd + t] = (f16)(acc * inv);
}

// ---------------------------------------------------------------------------
static void gemm(hipStream_t st, bool at,
                 const f16* A, int lda, long sAn, long sAh,
                 const f16* B, int ldb, long sBn, long sBh,
                 const float* bias, void* C, int om, int ldc, long sCn, long sCh,
                 const float* rs, long sRn, long sRh,
                 int Rr, int Kk, int Ncols, int Hdiv, int nb, int flags)
{
  dim3 g(Ncols / 64, Rr / 64, nb), b(256, 1, 1);
  if (at)
    k_gemm<true><<<g, b, 0, st>>>(A, lda, sAn, sAh, B, ldb, sBn, sBh, bias, C, om, ldc,
                                  sCn, sCh, rs, sRn, sRh, Kk, Hdiv, flags);
  else
    k_gemm<false><<<g, b, 0, st>>>(A, lda, sAn, sAh, B, ldb, sBn, sBh, bias, C, om, ldc,
                                   sCn, sCh, rs, sRn, sRh, Kk, Hdiv, flags);
}

extern "C" void kernel_launch(void* const* d_in, const int* in_sizes, int n_in,
                              void* d_out, int out_size, void* d_ws, size_t ws_size,
                              hipStream_t stream)
{
  (void)in_sizes; (void)n_in; (void)out_size; (void)ws_size;
  const float* expr     = (const float*)d_in[0];
  const float* coords   = (const float*)d_in[1];
  const float* gene_emb = (const float*)d_in[2];
  const float* gene_pos = (const float*)d_in[3];
  const float* val_w    = (const float*)d_in[4];
  const float* val_b    = (const float*)d_in[5];
  const float* val_g    = (const float*)d_in[6];
  const float* val_be   = (const float*)d_in[7];
  const float* sp_w     = (const float*)d_in[8];
  const float* sp_b     = (const float*)d_in[9];
  const float* sp_g     = (const float*)d_in[10];
  const float* sp_be    = (const float*)d_in[11];
  const float* qw_f = (const float*)d_in[44]; const float* qb_f = (const float*)d_in[45];
  const float* kw_f = (const float*)d_in[46]; const float* kb_f = (const float*)d_in[47];
  const float* vw_f = (const float*)d_in[48]; const float* vb_f = (const float*)d_in[49];
  const float* ow_f = (const float*)d_in[50]; const float* ob_f = (const float*)d_in[51];
  const float* proj = (const float*)d_in[52];

  // ---- workspace bump allocator ----
  char* p = (char*)d_ws;
  auto alloc = [&](size_t bytes) -> void* {
    void* r = (void*)p; p += (bytes + 255) & ~(size_t)255; return r;
  };
  f16 *wq16[DEPTHn], *wk16[DEPTHn], *wv16[DEPTHn], *wo16[DEPTHn];
  f16 *w1_16[DEPTHn], *w2_16[DEPTHn], *pmT16[DEPTHn];
  for (int L = 0; L < DEPTHn; ++L) {
    wq16[L] = (f16*)alloc(Dd * Dd * 2);  wk16[L] = (f16*)alloc(Dd * Dd * 2);
    wv16[L] = (f16*)alloc(Dd * Dd * 2);  wo16[L] = (f16*)alloc(Dd * Dd * 2);
    w1_16[L] = (f16*)alloc(Dd * FFf * 2); w2_16[L] = (f16*)alloc(FFf * Dd * 2);
    pmT16[L] = (f16*)alloc(DHd * Mp * 2);
  }
  f16* kw16 = (f16*)alloc(Dd * Dd * 2);
  f16* vw16 = (f16*)alloc(Dd * Dd * 2);
  f16* ow16 = (f16*)alloc(Dd * Dd * 2);
  float* valsb   = (float*)alloc((size_t)Nn * TOPKk * 4);
  int*   idxb    = (int*)alloc((size_t)Nn * TOPKk * 4);
  float* xbuf    = (float*)alloc((size_t)RCC * Dd * 4);
  f16*   h16     = (f16*)alloc((size_t)RCC * Dd * 2);
  f16*   q16     = (f16*)alloc((size_t)RCC * Dd * 2);
  f16*   k16     = (f16*)alloc((size_t)RCC * Dd * 2);
  f16*   v16     = (f16*)alloc((size_t)RCC * Dd * 2);
  f16*   ao16    = (f16*)alloc((size_t)RCC * Dd * 2);
  f16*   ffn16   = (f16*)alloc((size_t)RCC * FFf * 2);
  float* xdf     = (float*)alloc((size_t)Hh * RCC * Mp * 4);
  f16*   qp16    = (f16*)alloc((size_t)Hh * RCC * Mp * 2);
  f16*   kp16    = (f16*)alloc((size_t)Hh * RCC * Mp * 2);
  float* diagq   = (float*)alloc((size_t)Hh * RCC * 4);
  float* diagk   = (float*)alloc((size_t)Hh * RCC * 4);
  float* stabq   = (float*)alloc((size_t)Hh * RCC * 4);
  float* stabk   = (float*)alloc((size_t)NCC * Hh * 4);
  float* kpsum   = (float*)alloc((size_t)NCC * Hh * Mp * 4);
  float* dinv    = (float*)alloc((size_t)Hh * RCC * 4);
  f16*   ctx16   = (f16*)alloc((size_t)NCC * Hh * Mp * DHd * 2);
  float* qpool   = (float*)alloc((size_t)Nn * Dd * 4);
  f16*   pooled16 = (f16*)alloc((size_t)Nn * Dd * 2);

  // ---- weight conversion (once) ----
  auto cvt = [&](const float* s, f16* d, long n) {
    k_cvt<<<dim3((unsigned)((n + 255) / 256)), dim3(256), 0, stream>>>(s, d, n);
  };
  for (int L = 0; L < DEPTHn; ++L) {
    const int LB = 12 + L * 16;
    cvt((const float*)d_in[LB + 2],  wq16[L], Dd * Dd);
    cvt((const float*)d_in[LB + 4],  wk16[L], Dd * Dd);
    cvt((const float*)d_in[LB + 6],  wv16[L], Dd * Dd);
    cvt((const float*)d_in[LB + 8],  wo16[L], Dd * Dd);
    cvt((const float*)d_in[LB + 12], w1_16[L], Dd * FFf);
    cvt((const float*)d_in[LB + 14], w2_16[L], FFf * Dd);
    k_pmT<<<dim3((DHd * Mp + 255) / 256), dim3(256), 0, stream>>>(proj + (long)L * Mm * DHd,
                                                                  pmT16[L]);
  }
  cvt(kw_f, kw16, Dd * Dd);
  cvt(vw_f, vw16, Dd * Dd);
  cvt(ow_f, ow16, Dd * Dd);

  // ---- per-chunk pipeline (cells are independent end-to-end) ----
  for (int ch = 0; ch < Nn / NCC; ++ch) {
    const int cb = ch * NCC;
    k_topk<<<dim3(NCC), dim3(256), 0, stream>>>(expr, valsb, idxb, cb);
    k_embed<<<dim3(SPAD, NCC), dim3(256), 0, stream>>>(coords, sp_w, sp_b, sp_g, sp_be,
        val_w, val_b, val_g, val_be, gene_emb, gene_pos, valsb, idxb, xbuf, cb);

    for (int L = 0; L < DEPTHn; ++L) {
      const int LB = 12 + L * 16;
      const float* ln1g = (const float*)d_in[LB + 0];
      const float* ln1b = (const float*)d_in[LB + 1];
      const float* bq   = (const float*)d_in[LB + 3];
      const float* bk   = (const float*)d_in[LB + 5];
      const float* bv   = (const float*)d_in[LB + 7];
      const float* bo   = (const float*)d_in[LB + 9];
      const float* ln2g = (const float*)d_in[LB + 10];
      const float* ln2b = (const float*)d_in[LB + 11];
      const float* b1   = (const float*)d_in[LB + 13];
      const float* b2   = (const float*)d_in[LB + 15];

      k_ln<<<dim3(RCC), dim3(256), 0, stream>>>(xbuf, ln1g, ln1b, h16);
      gemm(stream, false, h16, Dd, 0, 0, wq16[L], Dd, 0, 0, bq, q16, OM_F16, Dd, 0, 0,
           nullptr, 0, 0, RCC, Dd, Dd, 1, 1, FG_ZPAD);
      gemm(stream, false, h16, Dd, 0, 0, wk16[L], Dd, 0, 0, bk, k16, OM_F16, Dd, 0, 0,
           nullptr, 0, 0, RCC, Dd, Dd, 1, 1, FG_ZPAD);
      gemm(stream, false, h16, Dd, 0, 0, wv16[L], Dd, 0, 0, bv, v16, OM_F16, Dd, 0, 0,
           nullptr, 0, 0, RCC, Dd, Dd, 1, 1, FG_ZPAD);
      k_diag<<<dim3(RCC), dim3(256), 0, stream>>>(q16, k16, diagq, diagk);

      // xd_q = q @ pmT (batched over heads), then stabilize + exp -> qp
      gemm(stream, false, q16, Dd, 0, DHd, pmT16[L], Mp, 0, 0, nullptr,
           xdf, OM_F32, Mp, 0, (long)RCC * Mp, nullptr, 0, 0, RCC, DHd, Mp, Hh, Hh, 0);
      k_rowmax<<<dim3(RCC, Hh), dim3(256), 0, stream>>>(xdf, stabq);
      k_exp<<<dim3(RCC, Hh), dim3(256), 0, stream>>>(xdf, diagq, stabq, qp16, 1);

      // xd_k -> kp (global per-head stabilizer, padded rows masked)
      gemm(stream, false, k16, Dd, 0, DHd, pmT16[L], Mp, 0, 0, nullptr,
           xdf, OM_F32, Mp, 0, (long)RCC * Mp, nullptr, 0, 0, RCC, DHd, Mp, Hh, Hh, 0);
      k_headmax<<<dim3(NCC, Hh), dim3(256), 0, stream>>>(xdf, stabk);
      k_exp<<<dim3(RCC, Hh), dim3(256), 0, stream>>>(xdf, diagk, stabk, kp16, 0);
      k_kpsum<<<dim3(NCC, Hh), dim3(256), 0, stream>>>(kp16, kpsum);

      // ctx[m,e] = sum_s kp[s,m] v[s,e]  (A-transposed WMMA GEMM, batched n,h)
      gemm(stream, true, kp16, Mp, (long)SPAD * Mp, (long)RCC * Mp,
           v16, Dd, (long)SPAD * Dd, DHd, nullptr,
           ctx16, OM_F16, DHd, (long)Hh * Mp * DHd, (long)Mp * DHd,
           nullptr, 0, 0, Mp, SPAD, DHd, Hh, NCC * Hh, 0);
      k_dinv<<<dim3(RCC), dim3(256), 0, stream>>>(qp16, kpsum, dinv);

      // out[s,e] = dinv[s] * (qp @ ctx)  -> ao16 interleaved [s][h*64+e]
      gemm(stream, false, qp16, Mp, (long)SPAD * Mp, (long)RCC * Mp,
           ctx16, DHd, (long)Hh * Mp * DHd, (long)Mp * DHd, nullptr,
           ao16, OM_F16, Dd, (long)SPAD * Dd, DHd,
           dinv, SPAD, RCC, SPAD, Mp, DHd, Hh, NCC * Hh, 0);

      // x += ao @ wo + bo   (residual accumulate)
      gemm(stream, false, ao16, Dd, 0, 0, wo16[L], Dd, 0, 0, bo, xbuf, OM_ACC, Dd, 0, 0,
           nullptr, 0, 0, RCC, Dd, Dd, 1, 1, 0);

      // FFN: x += gelu(ln2(x) @ w1 + b1) @ w2 + b2
      k_ln<<<dim3(RCC), dim3(256), 0, stream>>>(xbuf, ln2g, ln2b, h16);
      gemm(stream, false, h16, Dd, 0, 0, w1_16[L], FFf, 0, 0, b1, ffn16, OM_F16, FFf, 0, 0,
           nullptr, 0, 0, RCC, Dd, FFf, 1, 1, FG_GELU);
      gemm(stream, false, ffn16, FFf, 0, 0, w2_16[L], Dd, 0, 0, b2, xbuf, OM_ACC, Dd, 0, 0,
           nullptr, 0, 0, RCC, FFf, Dd, 1, 1, 0);
    }

    // final pooling (per chunk): k = gene@k_w+k_b, v = gene@v_w+v_b
    k_tof16<<<dim3((RCC * Dd + 255) / 256), dim3(256), 0, stream>>>(xbuf, h16,
                                                                    (long)RCC * Dd);
    gemm(stream, false, h16, Dd, 0, 0, kw16, Dd, 0, 0, kb_f, q16, OM_F16, Dd, 0, 0,
         nullptr, 0, 0, RCC, Dd, Dd, 1, 1, 0);
    gemm(stream, false, h16, Dd, 0, 0, vw16, Dd, 0, 0, vb_f, k16, OM_F16, Dd, 0, 0,
         nullptr, 0, 0, RCC, Dd, Dd, 1, 1, 0);
    k_qpool<<<dim3(NCC), dim3(256), 0, stream>>>(xbuf, qw_f, qb_f, qpool, cb);
    k_pool<<<dim3(NCC), dim3(256), 0, stream>>>(q16, k16, qpool, pooled16, cb);
  }

  // out = pooled @ o_w + o_b  (fp32 result)
  gemm(stream, false, pooled16, Dd, 0, 0, ow16, Dd, 0, 0, ob_f, d_out, OM_F32, Dd, 0, 0,
       nullptr, 0, 0, Nn, Dd, Dd, 1, 1, 0);
}
